// BilinearAttention_7189775254190
// MI455X (gfx1250) — matrix operand order repaired and checked by reading, not compiled, not run on hardware
//
#include <hip/hip_runtime.h>

// Additive (Bahdanau) attention, fused for MI455X (gfx1250, wave32, WMMA).
// B=4, TQ=128, TK=512, H=DQ=DK=DV=1024.
// d_in: query, key, value, Wq, bq, Wk, v (all f32)
// d_out: h [4*128*1024] then attn [4*128*512], f32
//
// Strategy: one-time f16 conversion (+ transpose of the B operands) so the
// WMMA GEMM inner loops are pure b128 loads + v_wmma, no per-iteration cvts
// or strided b32 gathers. Scores+softmax fused (never materializes the
// [B,TQ,TK,H] tensor), tanh via v_exp_f32/v_rcp_f32.

typedef __attribute__((ext_vector_type(16))) _Float16 v16h;
typedef __attribute__((ext_vector_type(8)))  _Float16 v8h;
typedef __attribute__((ext_vector_type(8)))  float    v8f;

#define H_DIM 1024
#define TQ 128
#define TK 512
#define NB 4

// ---------------------------------------------------------------------------
// Elementwise f32 -> f16 (layout preserved). n % 4 == 0.
// ---------------------------------------------------------------------------
__global__ __launch_bounds__(256)
void cvt_f16(const float* __restrict__ src, _Float16* __restrict__ dst, int n4) {
  int i = blockIdx.x * blockDim.x + threadIdx.x;
  if (i >= n4) return;
  const float4 f = ((const float4*)src)[i];
  v8h* d = (v8h*)dst;                // store 4 halves = 8 bytes via half-vec
  _Float16 h0 = (_Float16)f.x, h1 = (_Float16)f.y,
           h2 = (_Float16)f.z, h3 = (_Float16)f.w;
  ((__attribute__((ext_vector_type(4))) _Float16*)dst)[i] =
      (__attribute__((ext_vector_type(4))) _Float16){h0, h1, h2, h3};
  (void)d;
}

// ---------------------------------------------------------------------------
// Batched transpose + convert: src f32 [R,C] -> dst f16 [C,R].
// 32x32 LDS tile, 32x8 threads. Grid: (C/32, R/32, batches).
// ---------------------------------------------------------------------------
__global__ __launch_bounds__(256)
void transpose_cvt_f16(const float* __restrict__ src, _Float16* __restrict__ dst,
                       int R, int C, size_t sIn, size_t sOut) {
  __shared__ float tile[32][33];
  const float*   s = src + sIn  * blockIdx.z;
  _Float16*      d = dst + sOut * blockIdx.z;
  const int tx = threadIdx.x & 31;
  const int ty = threadIdx.x >> 5;          // 0..7
  const int c0 = blockIdx.x << 5;
  const int r0 = blockIdx.y << 5;
#pragma unroll
  for (int i = 0; i < 4; ++i) {
    tile[ty + 8 * i][tx] = s[(size_t)(r0 + ty + 8 * i) * C + c0 + tx];
  }
  __syncthreads();
#pragma unroll
  for (int i = 0; i < 4; ++i) {
    d[(size_t)(c0 + ty + 8 * i) * R + r0 + tx] = (_Float16)tile[tx][ty + 8 * i];
  }
}

// ---------------------------------------------------------------------------
// Batched WMMA GEMM: C[b] = A[b] (MxK, f16 row-major) * Bt[b]^T + bias.
// Bt is f16 [N,K] (pre-transposed), so every fragment load is a contiguous
// 16-byte b128. Each wave owns a 16x64 strip: A fragment reused by 4 WMMAs.
// M % 16 == 0, N % 64 == 0, K % 32 == 0.
// ---------------------------------------------------------------------------
__global__ __launch_bounds__(256)
void wmma_gemm_f16(const _Float16* __restrict__ A, const _Float16* __restrict__ Bt,
                   const float* __restrict__ bias, float* __restrict__ C,
                   int M, int N, int K,
                   size_t strideA, size_t strideB, size_t strideC) {
  const int lane = threadIdx.x & 31;
  const int wave = threadIdx.x >> 5;
  const int b    = blockIdx.y;

  const int strips_n = N >> 6;                         // 64-wide strips
  const int total    = (M >> 4) * strips_n;
  const int t = blockIdx.x * (blockDim.x >> 5) + wave; // uniform per wave
  if (t >= total) return;                              // wave-uniform exit

  const int tm = t / strips_n;
  const int tn = t - tm * strips_n;
  const int m0 = tm << 4;
  const int n0 = tn << 6;

  const int row = lane & 15;   // A: row within tile / B,D: column within tile
  const int hi  = lane >> 4;   // lane half-group

  const _Float16* __restrict__ Ab = A  + strideA * b;
  const _Float16* __restrict__ Bb = Bt + strideB * b;
  float*          __restrict__ Cb = C  + strideC * b;

  // A frag: lane holds row (m0+row), K elems {kb+hi*8..+7} and {kb+16+hi*8..+7}
  const _Float16* arow = Ab + (size_t)(m0 + row) * K;
  // B frag: lane holds column (n0 + c*16 + row) = row of Bt; K elems
  // {kb+hi*8..+7} and {kb+16+hi*8..+7} (same order as A).
  const _Float16* brow = Bb + (size_t)(n0 + row) * K;

  v8f acc0 = {}, acc1 = {}, acc2 = {}, acc3 = {};

  for (int kb = 0; kb < K; kb += 32) {
    const v8h a_lo = *(const v8h*)(arow + kb + hi * 8);
    const v8h a_hi = *(const v8h*)(arow + kb + hi * 8 + 16);
    const v16h a = __builtin_shufflevector(a_lo, a_hi,
        0, 1, 2, 3, 4, 5, 6, 7, 8, 9, 10, 11, 12, 13, 14, 15);
#pragma unroll
    for (int c = 0; c < 4; ++c) {
      const _Float16* pc = brow + (size_t)c * 16 * K + kb + hi * 8;
      const v8h b_lo = *(const v8h*)pc;
      const v8h b_hi = *(const v8h*)(pc + 16);
      const v16h bf = __builtin_shufflevector(b_lo, b_hi,
          0, 1, 2, 3, 4, 5, 6, 7, 8, 9, 10, 11, 12, 13, 14, 15);
      v8f& acc = (c == 0) ? acc0 : (c == 1) ? acc1 : (c == 2) ? acc2 : acc3;
      acc = __builtin_amdgcn_wmma_f32_16x16x32_f16(
          /*neg_a=*/false, a, /*neg_b=*/false, bf,
          /*c_mod=*/(short)0, acc, /*reuse_a=*/false, /*reuse_b=*/false);
    }
  }

  // D element r -> row m0 + r + 8*hi, col n0 + c*16 + row
#pragma unroll
  for (int c = 0; c < 4; ++c) {
    const v8f& acc = (c == 0) ? acc0 : (c == 1) ? acc1 : (c == 2) ? acc2 : acc3;
    const int col = n0 + c * 16 + row;
    const float bv = bias ? bias[col] : 0.0f;
#pragma unroll
    for (int r = 0; r < 8; ++r) {
      Cb[(size_t)(m0 + r + 8 * hi) * N + col] = acc[r] + bv;
    }
  }
}

// ---------------------------------------------------------------------------
// Fused scores + softmax. One block per (b,q); wq row + v in LDS, uh streamed.
// score[k] = sum_h v[h]*tanh(wq[h]+uh[k,h]); softmax over k.
// Writes attn (f32, output) and attn16 (f16 staging for the final GEMM).
// ---------------------------------------------------------------------------
__device__ __forceinline__ float fast_tanh(float x) {
  float e = __builtin_amdgcn_exp2f(x * 2.8853900817779268f);  // exp(2x)
  return 1.0f - 2.0f * __builtin_amdgcn_rcpf(e + 1.0f);
}

__global__ __launch_bounds__(256)
void scores_softmax(const float* __restrict__ wq, const float* __restrict__ uh,
                    const float* __restrict__ v, float* __restrict__ attn,
                    _Float16* __restrict__ attn16) {
  const int bq = blockIdx.x;      // b*TQ + q
  const int b  = bq >> 7;         // TQ = 128

  __shared__ float s_wq[H_DIM];
  __shared__ float s_v[H_DIM];
  __shared__ float s_sc[TK];
  __shared__ float red[8];

  const float* wqrow = wq + (size_t)bq * H_DIM;
  for (int h = threadIdx.x; h < H_DIM; h += blockDim.x) {
    s_wq[h] = wqrow[h];
    s_v[h]  = v[h];
  }
  __syncthreads();

  const int lane = threadIdx.x & 31;
  const int wave = threadIdx.x >> 5;
  const float* uhb = uh + (size_t)b * TK * H_DIM;

  for (int k = wave; k < TK; k += 8) {
    const float* urow = uhb + (size_t)k * H_DIM;
    float acc = 0.0f;
#pragma unroll 4
    for (int h = lane; h < H_DIM; h += 32) {
      acc = fmaf(s_v[h], fast_tanh(s_wq[h] + urow[h]), acc);
    }
#pragma unroll
    for (int off = 16; off; off >>= 1) acc += __shfl_xor(acc, off, 32);
    if (lane == 0) s_sc[k] = acc;
  }
  __syncthreads();

  float m = -3.4e38f;
  for (int k = threadIdx.x; k < TK; k += 256) m = fmaxf(m, s_sc[k]);
#pragma unroll
  for (int off = 16; off; off >>= 1) m = fmaxf(m, __shfl_xor(m, off, 32));
  if (lane == 0) red[wave] = m;
  __syncthreads();
  m = red[0];
#pragma unroll
  for (int w = 1; w < 8; ++w) m = fmaxf(m, red[w]);
  __syncthreads();

  float s = 0.0f;
  for (int k = threadIdx.x; k < TK; k += 256) {
    float e = __builtin_amdgcn_exp2f((s_sc[k] - m) * 1.4426950408889634f);
    s_sc[k] = e;
    s += e;
  }
#pragma unroll
  for (int off = 16; off; off >>= 1) s += __shfl_xor(s, off, 32);
  if (lane == 0) red[wave] = s;
  __syncthreads();
  s = 0.0f;
#pragma unroll
  for (int w = 0; w < 8; ++w) s += red[w];
  const float inv = __builtin_amdgcn_rcpf(s);

  float*    arow = attn   + (size_t)bq * TK;
  _Float16* hrow = attn16 + (size_t)bq * TK;
  for (int k = threadIdx.x; k < TK; k += 256) {
    const float p = s_sc[k] * inv;
    arow[k] = p;
    hrow[k] = (_Float16)p;
  }
}

// ---------------------------------------------------------------------------
extern "C" void kernel_launch(void* const* d_in, const int* in_sizes, int n_in,
                              void* d_out, int out_size, void* d_ws, size_t ws_size,
                              hipStream_t stream) {
  const float* query = (const float*)d_in[0];  // [4,128,1024]
  const float* key   = (const float*)d_in[1];  // [4,512,1024]
  const float* value = (const float*)d_in[2];  // [4,512,1024]
  const float* Wq    = (const float*)d_in[3];  // [1024,1024]
  const float* bq    = (const float*)d_in[4];  // [1024]
  const float* Wk    = (const float*)d_in[5];  // [1024,1024]
  const float* vvec  = (const float*)d_in[6];  // [1024]

  float* out_h    = (float*)d_out;                       // [4,128,1024]
  float* out_attn = out_h + (size_t)NB * TQ * H_DIM;     // [4,128,512]

  // ---- workspace layout ----
  float* ws_wq = (float*)d_ws;                           // f32 [512,1024]   2 MB
  float* ws_uh = ws_wq + (size_t)NB * TQ * H_DIM;        // f32 [2048,1024]  8 MB
  _Float16* hbase = (_Float16*)(ws_uh + (size_t)NB * TK * H_DIM);
  _Float16* q16    = hbase;                              // [512,1024]   1 MB
  _Float16* k16    = q16  + (size_t)NB * TQ * H_DIM;     // [2048,1024]  4 MB
  _Float16* Wqt    = k16  + (size_t)NB * TK * H_DIM;     // [1024,1024]  2 MB
  _Float16* Wkt    = Wqt  + (size_t)H_DIM * H_DIM;       // [1024,1024]  2 MB
  _Float16* vt     = Wkt  + (size_t)H_DIM * H_DIM;       // [4][1024,512] 4 MB
  _Float16* attn16 = vt   + (size_t)NB * H_DIM * TK;     // [4][128,512] .5 MB

  // ---- one-time f16 conversions / transposes (L2-resident, ~1 us) ----
  cvt_f16<<<(NB * TQ * H_DIM / 4 + 255) / 256, 256, 0, stream>>>(
      query, q16, NB * TQ * H_DIM / 4);
  cvt_f16<<<(NB * TK * H_DIM / 4 + 255) / 256, 256, 0, stream>>>(
      key, k16, NB * TK * H_DIM / 4);
  {
    dim3 g(H_DIM / 32, H_DIM / 32, 1);
    transpose_cvt_f16<<<g, 256, 0, stream>>>(Wq, Wqt, H_DIM, H_DIM, 0, 0);
    transpose_cvt_f16<<<g, 256, 0, stream>>>(Wk, Wkt, H_DIM, H_DIM, 0, 0);
  }
  {
    dim3 g(H_DIM / 32, TK / 32, NB);  // value [512,1024] -> vt [1024,512]
    transpose_cvt_f16<<<g, 256, 0, stream>>>(value, vt, TK, H_DIM,
                                             (size_t)TK * H_DIM,
                                             (size_t)H_DIM * TK);
  }

  // 1) wq = query @ Wq + bq  (M=512, N=1024, K=1024): 512 strips -> 64 blocks
  wmma_gemm_f16<<<dim3(512 / 8, 1), 256, 0, stream>>>(
      q16, Wqt, bq, ws_wq, NB * TQ, H_DIM, H_DIM, 0, 0, 0);

  // 2) uh = key @ Wk  (M=2048, N=1024, K=1024): 2048 strips -> 256 blocks
  wmma_gemm_f16<<<dim3(2048 / 8, 1), 256, 0, stream>>>(
      k16, Wkt, nullptr, ws_uh, NB * TK, H_DIM, H_DIM, 0, 0, 0);

  // 3) fused scores + softmax -> attn (f32 output) + attn16 (f16 staging)
  scores_softmax<<<NB * TQ, 256, 0, stream>>>(ws_wq, ws_uh, vvec,
                                              out_attn, attn16);

  // 4) h = attn @ value  (batched: M=128, N=1024, K=512): 128 strips -> 16 blocks
  wmma_gemm_f16<<<dim3(128 / 8, NB), 256, 0, stream>>>(
      attn16, vt, nullptr, out_h, TQ, H_DIM, TK,
      (size_t)TQ * TK, (size_t)H_DIM * TK, (size_t)TQ * H_DIM);
}
